// FunctionalMambaMultiLayerST_14405320311412
// MI455X (gfx1250) — compile-verified, matrix operand order checked by reading, not run on hardware
//
#include <hip/hip_runtime.h>
#include <hip/hip_bf16.h>

// ---------------------------------------------------------------------------
// Problem constants (from reference)
// ---------------------------------------------------------------------------
#define BB     4
#define HH     32
#define WW     32
#define CC     256
#define LL     (HH * WW)          // 1024
#define TOK    (BB * LL)          // 4096 tokens (also 8 windows * 512 seq)
#define DI     512                // d_inner
#define DS     16                 // d_state
#define DCONV  4
#define DTR    16                 // dt_rank
#define XPN    (DTR + 2 * DS)     // 48
#define LSEQ   512                // seq length per window
#define NWIN   8                  // window-batch (B * 2)
#define EPSLN  1e-5f

typedef __attribute__((ext_vector_type(16))) __bf16 v16bf;
typedef __attribute__((ext_vector_type(8)))  float  v8f;

union FragBF {
    uint4 q[2];
    v16bf v;
};

// ---------------------------------------------------------------------------
// Helpers
// ---------------------------------------------------------------------------
__device__ __forceinline__ unsigned short f2bf(float f) {
    union { float f; unsigned int u; } a;
    a.f = f;
    unsigned int u = a.u;
    unsigned int r = u + 0x7FFFu + ((u >> 16) & 1u);   // round-to-nearest-even
    return (unsigned short)(r >> 16);
}

__device__ __forceinline__ float siluf(float x) {
    return x / (1.0f + __expf(-x));
}

__device__ __forceinline__ float softplusf(float x) {
    return (x > 20.0f) ? x : log1pf(__expf(x));
}

// ---------------------------------------------------------------------------
// f32 -> bf16 bulk convert (weights repack)
// ---------------------------------------------------------------------------
__global__ __launch_bounds__(256) void cvt_bf16_kernel(
    const float* __restrict__ src, unsigned short* __restrict__ dst, int n) {
    int i = blockIdx.x * 256 + threadIdx.x;
    if (i < n) dst[i] = f2bf(src[i]);
}

// ---------------------------------------------------------------------------
// Residual add + LayerNorm over C=256. One block (256 thr) per token.
//   r = base[idx] + (add ? add[idx] : 0);  resid[idx] = r;  out = LN(r)*w + b
// ---------------------------------------------------------------------------
__global__ __launch_bounds__(256) void ln_kernel(
    const float* __restrict__ base, const float* __restrict__ add,
    float* __restrict__ resid, float* __restrict__ out,
    const float* __restrict__ w, const float* __restrict__ b) {
    __shared__ float sm[256];
    int tid = threadIdx.x;
    size_t idx = (size_t)blockIdx.x * CC + tid;

    float r = base[idx] + (add ? add[idx] : 0.0f);
    resid[idx] = r;

    sm[tid] = r;
    __syncthreads();
    for (int s = 128; s > 0; s >>= 1) {
        if (tid < s) sm[tid] += sm[tid + s];
        __syncthreads();
    }
    float mean = sm[0] * (1.0f / CC);
    __syncthreads();

    float d = r - mean;
    sm[tid] = d * d;
    __syncthreads();
    for (int s = 128; s > 0; s >>= 1) {
        if (tid < s) sm[tid] += sm[tid + s];
        __syncthreads();
    }
    float var = sm[0] * (1.0f / CC);

    out[idx] = d * rsqrtf(var + EPSLN) * w[tid] + b[tid];
}

// ---------------------------------------------------------------------------
// Swin roll + window-partition fused into a gather with f32->bf16 convert.
// Sequence token t = bp*512 + l  maps to spatial (b, h, w):
//   b = bp/2, s1 = bp&1
//   mix 0 (H windows): row = l/32, w = l%32, h = (2*row + s1 + 2) & 31
//   mix 1 (W windows): h = l/16, col = l%16, w = (2*col + s1 + 2) & 31
// ---------------------------------------------------------------------------
__device__ __forceinline__ size_t seq_to_spatial(int t, int c, int mix) {
    int bp = t >> 9;          // t / 512
    int l  = t & 511;
    int bb = bp >> 1;
    int s1 = bp & 1;
    int h, w;
    if (mix == 0) {
        int row = l >> 5;                 // l / 32
        w = l & 31;
        h = ((row << 1) + s1 + 2) & 31;
    } else {
        h = l >> 4;                       // l / 16
        int col = l & 15;
        w = ((col << 1) + s1 + 2) & 31;
    }
    return ((size_t)((bb * HH + h) * WW + w)) * CC + c;
}

__global__ __launch_bounds__(256) void gather_bf_kernel(
    const float* __restrict__ spat, unsigned short* __restrict__ dst, int mix) {
    int idx = blockIdx.x * 256 + threadIdx.x;   // over TOK*CC
    int t = idx >> 8;
    int c = idx & 255;
    dst[idx] = f2bf(spat[seq_to_spatial(t, c, mix)]);
}

__global__ __launch_bounds__(256) void scatter_f32_kernel(
    const float* __restrict__ seq, float* __restrict__ spat, int mix) {
    int idx = blockIdx.x * 256 + threadIdx.x;   // over TOK*CC
    int t = idx >> 8;
    int c = idx & 255;
    spat[seq_to_spatial(t, c, mix)] = seq[idx];
}

// ---------------------------------------------------------------------------
// bf16 WMMA GEMM:  C[M,N](f32) = A[M,K](bf16, row-major) * W[N,K](bf16)^T
// Block = 256 threads = 8 waves; each wave owns one 16x16 output tile.
// Grid: (M/128, N/16). K-loop steps 32 with v_wmma_f32_16x16x32_bf16.
// Fragment layout per CDNA5 ISA 7.12.2:
//   lanes 0-15 hold row r, K in {kk..kk+7} U {kk+16..kk+23}
//   lanes16-31 hold row r, K in {kk+8..kk+15} U {kk+24..kk+31}
// C layout: VGPR i -> M = mt + i + 8*(lane>=16), N = nt + (lane&15)
// ---------------------------------------------------------------------------
__global__ __launch_bounds__(256) void gemm_bf16_kernel(
    const unsigned short* __restrict__ A, const unsigned short* __restrict__ Wt,
    float* __restrict__ C, int M, int N, int K) {
    int wave = threadIdx.x >> 5;
    int lane = threadIdx.x & 31;
    int mt = blockIdx.x * 128 + wave * 16;
    int nt = blockIdx.y * 16;
    int r  = lane & 15;
    int hi = lane >> 4;

    const unsigned short* pa = A  + (size_t)(mt + r) * K + hi * 8;
    const unsigned short* pb = Wt + (size_t)(nt + r) * K + hi * 8;

    v8f acc = {};
    for (int kk = 0; kk < K; kk += 32) {
        FragBF fa, fb;
        const uint4* qa = (const uint4*)(pa + kk);
        const uint4* qb = (const uint4*)(pb + kk);
        fa.q[0] = qa[0];
        fa.q[1] = qa[2];     // +16 bf16 elements
        fb.q[0] = qb[0];
        fb.q[1] = qb[2];
        if (kk + 32 < K) {   // global_prefetch_b8 of next K tile
            __builtin_prefetch(pa + kk + 32, 0, 1);
            __builtin_prefetch(pb + kk + 32, 0, 1);
        }
        acc = __builtin_amdgcn_wmma_f32_16x16x32_bf16(
            /*neg_a=*/false, fa.v, /*neg_b=*/false, fb.v,
            /*c_mod=*/(short)0, acc, /*reuse_a=*/false, /*reuse_b=*/false);
    }

    float* co = C + (size_t)(mt + hi * 8) * N + nt + r;
#pragma unroll
    for (int i = 0; i < 8; ++i) co[(size_t)i * N] = acc[i];
}

// ---------------------------------------------------------------------------
// Causal depthwise conv1d (D_CONV=4) + bias + SiLU over the xh half of xz.
// xz layout: [t][1024], columns 0..511 = pre-conv xh, 512..1023 = z gate.
// One thread per (t, d). Emits f32 xh and bf16 copy for the x_proj GEMM.
// ---------------------------------------------------------------------------
__global__ __launch_bounds__(256) void conv_silu_kernel(
    const float* __restrict__ xz, const float* __restrict__ cw,
    const float* __restrict__ cb, float* __restrict__ xh,
    unsigned short* __restrict__ xh_bf) {
    int idx = blockIdx.x * 256 + threadIdx.x;   // over TOK*DI
    int t = idx >> 9;
    int d = idx & 511;
    int bp = t >> 9;
    int l  = t & 511;

    float acc = cb[d];
#pragma unroll
    for (int j = 0; j < DCONV; ++j) {
        int ll = l - (DCONV - 1) + j;
        if (ll >= 0)
            acc += cw[d * DCONV + j] * xz[(size_t)((bp << 9) + ll) * (2 * DI) + d];
    }
    float s = siluf(acc);
    xh[idx]    = s;
    xh_bf[idx] = f2bf(s);
}

// ---------------------------------------------------------------------------
// Fused selective scan. One thread per (window bp, channel d):
//   - dt_proj (K=16) + softplus -> delta
//   - h[n] = exp(delta*A[n])*h[n] + delta*B[t][n]*xh ;  y = <h, C[t]> + D*xh
//   - gate y *= silu(z), emit bf16 for out_proj GEMM
// dbc layout: [t][48] = [dt(16) | B(16) | C(16)]
// ---------------------------------------------------------------------------
__global__ __launch_bounds__(256) void scan_kernel(
    const float* __restrict__ xz, const float* __restrict__ xh,
    const float* __restrict__ dbc, const float* __restrict__ dtw,
    const float* __restrict__ dtb, const float* __restrict__ alog,
    const float* __restrict__ Dp, unsigned short* __restrict__ ybf) {
    int g  = blockIdx.x * 256 + threadIdx.x;    // over NWIN*DI = 4096
    int d  = g & 511;
    int bp = g >> 9;

    float a[DS], wdt[DTR], h[DS];
#pragma unroll
    for (int n = 0; n < DS; ++n) { a[n] = -__expf(alog[d * DS + n]); h[n] = 0.0f; }
#pragma unroll
    for (int r = 0; r < DTR; ++r) wdt[r] = dtw[d * DTR + r];
    float bdt = dtb[d];
    float Dd  = Dp[d];

    for (int l = 0; l < LSEQ; ++l) {
        int t = (bp << 9) + l;
        const float* row = dbc + (size_t)t * XPN;

        float dt = bdt;
#pragma unroll
        for (int r = 0; r < DTR; ++r) dt += row[r] * wdt[r];
        float delta = softplusf(dt);

        float xv = xh[(size_t)t * DI + d];
        float acc = 0.0f;
#pragma unroll
        for (int n = 0; n < DS; ++n) {
            float dA = __expf(delta * a[n]);
            h[n] = dA * h[n] + delta * row[DTR + n] * xv;
            acc += h[n] * row[DTR + DS + n];
        }
        float y = acc + Dd * xv;
        float z = xz[(size_t)t * (2 * DI) + DI + d];
        ybf[(size_t)t * DI + d] = f2bf(y * siluf(z));
    }
}

// ---------------------------------------------------------------------------
// Host-side launcher
// ---------------------------------------------------------------------------
extern "C" void kernel_launch(void* const* d_in, const int* in_sizes, int n_in,
                              void* d_out, int out_size, void* d_ws, size_t ws_size,
                              hipStream_t stream) {
    (void)in_sizes; (void)n_in; (void)out_size; (void)ws_size;

    const float* x          = (const float*)d_in[0];
    const float* in_proj_w  = (const float*)d_in[1];
    const float* conv_w     = (const float*)d_in[2];
    const float* conv_b     = (const float*)d_in[3];
    const float* x_proj_w   = (const float*)d_in[4];
    const float* dt_proj_w  = (const float*)d_in[5];
    const float* dt_proj_b  = (const float*)d_in[6];
    const float* A_log      = (const float*)d_in[7];
    const float* D_param    = (const float*)d_in[8];
    const float* out_proj_w = (const float*)d_in[9];
    const float* norm_w     = (const float*)d_in[10];
    const float* norm_b     = (const float*)d_in[11];
    const float* normf_w    = (const float*)d_in[12];
    const float* normf_b    = (const float*)d_in[13];
    float* out = (float*)d_out;

    // ---- workspace layout (~52 MB), 256B aligned slabs -------------------
    char* ws = (char*)d_ws;
    size_t off = 0;
    auto slab = [&](size_t bytes) -> char* {
        char* p = ws + off;
        off = (off + bytes + 255) & ~(size_t)255;
        return p;
    };
    float*          resid   = (float*)         slab((size_t)TOK * CC * 4);
    float*          spat    = (float*)         slab((size_t)TOK * CC * 4);
    float*          xz      = (float*)         slab((size_t)TOK * 2 * DI * 4);
    float*          xh      = (float*)         slab((size_t)TOK * DI * 4);
    float*          dbc     = (float*)         slab((size_t)TOK * XPN * 4);
    float*          outseq  = (float*)         slab((size_t)TOK * CC * 4);
    unsigned short* act_bf  = (unsigned short*)slab((size_t)TOK * CC * 2);
    unsigned short* xh_bf   = (unsigned short*)slab((size_t)TOK * DI * 2);
    unsigned short* y_bf    = (unsigned short*)slab((size_t)TOK * DI * 2);
    unsigned short* w_in_bf = (unsigned short*)slab((size_t)4 * 2 * DI * CC * 2);
    unsigned short* w_xp_bf = (unsigned short*)slab((size_t)4 * XPN * DI * 2);
    unsigned short* w_op_bf = (unsigned short*)slab((size_t)4 * CC * DI * 2);

    // ---- one-time weight repack f32 -> bf16 ------------------------------
    {
        int n0 = 4 * 2 * DI * CC;   // in_proj
        int n1 = 4 * XPN * DI;      // x_proj
        int n2 = 4 * CC * DI;       // out_proj
        cvt_bf16_kernel<<<(n0 + 255) / 256, 256, 0, stream>>>(in_proj_w, w_in_bf, n0);
        cvt_bf16_kernel<<<(n1 + 255) / 256, 256, 0, stream>>>(x_proj_w, w_xp_bf, n1);
        cvt_bf16_kernel<<<(n2 + 255) / 256, 256, 0, stream>>>(out_proj_w, w_op_bf, n2);
    }

    const int elem_blocks = (TOK * CC) / 256;       // 4096
    const int conv_blocks = (TOK * DI) / 256;       // 8192
    const int scan_blocks = (NWIN * DI) / 256;      // 16

    for (int l = 0; l < 2; ++l) {
        // residual += hidden ; LN -> spat
        ln_kernel<<<TOK, 256, 0, stream>>>(
            (l == 0) ? x : resid, (l == 0) ? nullptr : spat,
            resid, spat, norm_w + l * CC, norm_b + l * CC);

        for (int i = 0; i < 2; ++i) {
            int li = l * 2 + i;

            // roll + partition + bf16 convert
            gather_bf_kernel<<<elem_blocks, 256, 0, stream>>>(spat, act_bf, i);

            // in_proj: [4096,256] x [1024,256]^T -> xz
            gemm_bf16_kernel<<<dim3(TOK / 128, (2 * DI) / 16), 256, 0, stream>>>(
                act_bf, w_in_bf + (size_t)li * 2 * DI * CC, xz, TOK, 2 * DI, CC);

            // causal depthwise conv + SiLU
            conv_silu_kernel<<<conv_blocks, 256, 0, stream>>>(
                xz, conv_w + (size_t)li * DI * DCONV, conv_b + (size_t)li * DI,
                xh, xh_bf);

            // x_proj: [4096,512] x [48,512]^T -> dbc
            gemm_bf16_kernel<<<dim3(TOK / 128, XPN / 16), 256, 0, stream>>>(
                xh_bf, w_xp_bf + (size_t)li * XPN * DI, dbc, TOK, XPN, DI);

            // fused dt_proj + softplus + selective scan + gate -> y_bf
            scan_kernel<<<scan_blocks, 256, 0, stream>>>(
                xz, xh, dbc,
                dt_proj_w + (size_t)li * DI * DTR, dt_proj_b + (size_t)li * DI,
                A_log + (size_t)li * DI * DS, D_param + (size_t)li * DI, y_bf);

            // out_proj: [4096,512] x [256,512]^T -> outseq
            gemm_bf16_kernel<<<dim3(TOK / 128, CC / 16), 256, 0, stream>>>(
                y_bf, w_op_bf + (size_t)li * CC * DI, outseq, TOK, CC, DI);

            // reverse partition + unroll back into spatial layout
            scatter_f32_kernel<<<elem_blocks, 256, 0, stream>>>(outseq, spat, i);
        }
    }

    // final: residual += hidden ; LN_f -> d_out
    ln_kernel<<<TOK, 256, 0, stream>>>(resid, spat, resid, out, normf_w, normf_b);
}
